// SimpleRNN_70128226009289
// MI455X (gfx1250) — matrix operand order repaired and checked by reading, not compiled, not run on hardware
//
#include <hip/hip_runtime.h>
#include <math.h>

#define SEQ  2048
#define EMBD 1024
#define HID  1024
#define NWG  128     // workgroups in the persistent scan
#define RPW  8       // rows of W_hh per workgroup (128*8 = 1024)

typedef __attribute__((ext_vector_type(16))) _Float16 v16h;
typedef __attribute__((ext_vector_type(8)))  _Float16 v8h;
typedef __attribute__((ext_vector_type(4)))  _Float16 v4h;
typedef __attribute__((ext_vector_type(8)))  float    v8f;

// ---- workspace layout (bytes, 256B-aligned offsets) ----
#define OFF_CNT 0                                  // 1 x u32 barrier counter
#define OFF_H   256                                // 2 x 1024 f32 (double-buffered h)
#define OFF_W16 16384                              // 1024x1024 f16  (W_ih)
#define OFF_E16 (OFF_W16 + (size_t)HID * EMBD * 2) // 2048x1024 f16  (renormed e)
#define OFF_XH  (OFF_E16 + (size_t)SEQ * EMBD * 2) // 2048x1024 f32  (xh)

// ---------------- init: zero h double-buffer + barrier counter ----------------
__global__ void k_init(float* hbuf, unsigned* cnt) {
  int t = threadIdx.x;
  for (int i = t; i < 2 * HID; i += blockDim.x) hbuf[i] = 0.0f;
  if (t == 0) *cnt = 0u;
}

// ---------------- embedding gather + L2 max_norm + f16 convert ----------------
__global__ void __launch_bounds__(256) k_embed(const int* __restrict__ idx,
                                               const float* __restrict__ emb,
                                               _Float16* __restrict__ e16) {
  __shared__ float wsum[8];
  int s = blockIdx.x, t = threadIdx.x;
  int row = idx[s];
  float4 v = ((const float4*)(emb + (size_t)row * EMBD))[t];   // 256*4 = 1024
  float ss = v.x * v.x + v.y * v.y + v.z * v.z + v.w * v.w;
  for (int o = 16; o > 0; o >>= 1) ss += __shfl_xor(ss, o, 32);
  if ((t & 31) == 0) wsum[t >> 5] = ss;
  __syncthreads();
  if (t == 0) {
    float tot = 0.0f;
    for (int i = 0; i < 8; ++i) tot += wsum[i];
    float nrm = sqrtf(tot);
    wsum[0] = fminf(1.0f, 1.0f / fmaxf(nrm, 1e-7f));
  }
  __syncthreads();
  float sc = wsum[0];
  v4h o;
  o[0] = (_Float16)(v.x * sc); o[1] = (_Float16)(v.y * sc);
  o[2] = (_Float16)(v.z * sc); o[3] = (_Float16)(v.w * sc);
  *(v4h*)(e16 + (size_t)s * EMBD + t * 4) = o;
}

// ---------------- W_ih f32 -> f16 ----------------
__global__ void k_w2h(const float* __restrict__ W, _Float16* __restrict__ W16) {
  int i = blockIdx.x * blockDim.x + threadIdx.x;          // covers (HID*EMBD)/4
  float4 v = ((const float4*)W)[i];
  v4h o;
  o[0] = (_Float16)v.x; o[1] = (_Float16)v.y;
  o[2] = (_Float16)v.z; o[3] = (_Float16)v.w;
  *(v4h*)(W16 + (size_t)i * 4) = o;
}

// ---------------- xh = e @ W_ih^T + b_ih via v_wmma_f32_16x16x32_f16 ----------
__global__ void __launch_bounds__(256)
k_gemm_xh(const _Float16* __restrict__ E, const _Float16* __restrict__ W,
          const float* __restrict__ b_ih, float* __restrict__ XH) {
  const int NT = HID / 16;                                   // 64 N-tiles
  int wave = blockIdx.x * (blockDim.x >> 5) + (threadIdx.x >> 5);
  int lane = threadIdx.x & 31;
  int mt = wave / NT, nt = wave % NT;
  int m0 = mt * 16, n0 = nt * 16;
  int ln = lane & 15, kh = lane >> 4;

  const _Float16* arow = E + (size_t)(m0 + ln) * EMBD;       // A row m
  const _Float16* brow = W + (size_t)(n0 + ln) * EMBD;       // B col n = W_ih row n
  v8f c = {};
  for (int k0 = 0; k0 < EMBD; k0 += 32) {
    v8h alo = *(const v8h*)(arow + k0 + kh * 8);             // K = kh*8 + [0..7]
    v8h ahi = *(const v8h*)(arow + k0 + 16 + kh * 8);        // K = 16+kh*8+[0..7]
    v16h a = __builtin_shufflevector(alo, ahi,
                                     0, 1, 2, 3, 4, 5, 6, 7,
                                     8, 9, 10, 11, 12, 13, 14, 15);
    v8h blo = *(const v8h*)(brow + k0 + kh * 8);             // K = kh*8 + [0..7]
    v8h bhi = *(const v8h*)(brow + k0 + 16 + kh * 8);        // K = 16+kh*8+[0..7]
    v16h b = __builtin_shufflevector(blo, bhi,
                                     0, 1, 2, 3, 4, 5, 6, 7,
                                     8, 9, 10, 11, 12, 13, 14, 15);
    c = __builtin_amdgcn_wmma_f32_16x16x32_f16(false, a, false, b,
                                               (short)0, c, false, false);
  }
  float bias = b_ih[n0 + ln];
  int mb = m0 + kh * 8;                                      // D: VGPR r -> M = mb + r
#pragma unroll
  for (int r = 0; r < 8; ++r)
    XH[(size_t)(mb + r) * HID + n0 + ln] = c[r] + bias;
}

// ---------------- persistent recurrent scan ----------------
// 128 WGs; WG w owns rows [w*8, w*8+8) of W_hh, pinned in LDS (stride 1025 pads banks).
__global__ void __launch_bounds__(256)
k_scan(const float* __restrict__ W_hh, const float* __restrict__ b_hh,
       const float* __restrict__ XH, float* hbuf, unsigned* cnt) {
  __shared__ float Wsl[RPW * 1025];
  __shared__ __align__(16) float hb[HID];
  __shared__ float acc[RPW];
  int t = threadIdx.x, w = blockIdx.x;
  int r = t & 7, p = t >> 3;                       // row-in-slice, 32 K-parts

  for (int i = t; i < RPW * HID; i += 256) {       // stage W_hh slice once
    int rr = i >> 10, kk = i & 1023;
    Wsl[rr * 1025 + kk] = W_hh[(size_t)(w * RPW + rr) * HID + kk];
  }
  float myb = (t < RPW) ? b_hh[w * RPW + t] : 0.0f;
  __syncthreads();

  for (int step = 0; step < SEQ; ++step) {
    const float* hcur = hbuf + (step & 1) * HID;
    float* hnxt = hbuf + ((step + 1) & 1) * HID;

    ((float4*)hb)[t] = ((const float4*)hcur)[t];   // 4 KB h -> LDS
    if (t < RPW) acc[t] = 0.0f;
    __syncthreads();

    const float* wr = Wsl + r * 1025 + p * 32;
    const float* hp = hb + p * 32;
    float partial = 0.0f;
#pragma unroll 8
    for (int i = 0; i < 32; ++i) partial = fmaf(wr[i], hp[i], partial);
    partial += __shfl_xor(partial, 8, 32);         // fold the 4 parts per wave
    partial += __shfl_xor(partial, 16, 32);
    if ((t & 31) < 8) atomicAdd(&acc[r], partial); // ds_add_f32 across 8 waves
    __syncthreads();

    if (t < RPW) {
      int g = w * RPW + t;
      hnxt[g] = tanhf(XH[(size_t)step * HID + g] + myb + acc[t]);
    }
    __threadfence();                               // every thread drains its stores
    __syncthreads();
    if (t == 0) {                                  // device-wide sense barrier
      __hip_atomic_fetch_add(cnt, 1u, __ATOMIC_RELEASE, __HIP_MEMORY_SCOPE_AGENT);
      unsigned target = (unsigned)(NWG * (step + 1));
      while (__hip_atomic_load(cnt, __ATOMIC_ACQUIRE, __HIP_MEMORY_SCOPE_AGENT) < target)
        __builtin_amdgcn_s_sleep(1);
    }
    __syncthreads();
  }
}

// ---------------- head: logits + log_softmax ----------------
__global__ void __launch_bounds__(256)
k_head(const float* __restrict__ h, const float* __restrict__ W_ho,
       const float* __restrict__ b_ho, float* __restrict__ out) {
  __shared__ float red[3][8];
  int t = threadIdx.x;
  float pl[3] = {0.0f, 0.0f, 0.0f};
  for (int k = t; k < HID; k += 256) {
    float hv = h[k];
    pl[0] = fmaf(hv, W_ho[k], pl[0]);
    pl[1] = fmaf(hv, W_ho[HID + k], pl[1]);
    pl[2] = fmaf(hv, W_ho[2 * HID + k], pl[2]);
  }
#pragma unroll
  for (int o = 0; o < 3; ++o)
    for (int s = 16; s > 0; s >>= 1) pl[o] += __shfl_xor(pl[o], s, 32);
  if ((t & 31) == 0) {
    red[0][t >> 5] = pl[0]; red[1][t >> 5] = pl[1]; red[2][t >> 5] = pl[2];
  }
  __syncthreads();
  if (t == 0) {
    float l[3];
    for (int o = 0; o < 3; ++o) {
      float s = 0.0f;
      for (int i = 0; i < 8; ++i) s += red[o][i];
      l[o] = s + b_ho[o];
    }
    float m = fmaxf(l[0], fmaxf(l[1], l[2]));
    float lse = m + logf(expf(l[0] - m) + expf(l[1] - m) + expf(l[2] - m));
    out[0] = l[0] - lse; out[1] = l[1] - lse; out[2] = l[2] - lse;
  }
}

extern "C" void kernel_launch(void* const* d_in, const int* in_sizes, int n_in,
                              void* d_out, int out_size, void* d_ws, size_t ws_size,
                              hipStream_t stream) {
  (void)in_sizes; (void)n_in; (void)out_size; (void)ws_size;
  const int*   idx  = (const int*)d_in[0];
  const float* emb  = (const float*)d_in[1];
  const float* W_ih = (const float*)d_in[2];
  const float* b_ih = (const float*)d_in[3];
  const float* W_hh = (const float*)d_in[4];
  const float* b_hh = (const float*)d_in[5];
  const float* W_ho = (const float*)d_in[6];
  const float* b_ho = (const float*)d_in[7];
  float* out = (float*)d_out;

  char* ws = (char*)d_ws;
  unsigned* cnt  = (unsigned*)(ws + OFF_CNT);
  float*    hbuf = (float*)(ws + OFF_H);
  _Float16* W16  = (_Float16*)(ws + OFF_W16);
  _Float16* E16  = (_Float16*)(ws + OFF_E16);
  float*    XH   = (float*)(ws + OFF_XH);

  k_init<<<1, 256, 0, stream>>>(hbuf, cnt);
  k_embed<<<SEQ, 256, 0, stream>>>(idx, emb, E16);
  k_w2h<<<(HID * EMBD / 4) / 256, 256, 0, stream>>>(W_ih, W16);
  k_gemm_xh<<<(SEQ / 16) * (HID / 16) / 8, 256, 0, stream>>>(E16, W16, b_ih, XH);
  k_scan<<<NWG, 256, 0, stream>>>(W_hh, b_hh, XH, hbuf, cnt);
  k_head<<<1, 256, 0, stream>>>(hbuf, W_ho, b_ho, out);  // SEQ even -> final h in buf 0
}
